// DecoderTransformer_15023795601960
// MI455X (gfx1250) — matrix operand order repaired and checked by reading, not compiled, not run on hardware
//
#include <hip/hip_runtime.h>

// ---------------------------------------------------------------------------
// CDNA5 (gfx1250) decoder-transformer forward.
//   wave32, WMMA 16x16x32 bf16 -> f32 accumulate.
// Pipeline:
//   1) embed:  x = we[ids] + pe                       (f32 + bf16 copy)
//   2) cvtT:   Wq/Wk/Wv/proj_w/fc_w -> bf16 TRANSPOSED [N][K]
//              (fc_w^T bf16 = 131MB, fits 192MB L2; [n][k] rows let GEMM
//               stage B with b128 copies instead of b16 scatter)
//   3) gemm x3: q/k/v = x @ W[h]   (batched over heads, bf16 out)
//      transT:  v -> v^T [h][hs][t] (one-time, amortized over key-tile reuse)
//   4) flash attn: per-head online-softmax, WMMA QK^T and PV -> cat (bf16)
//   5) gemm:   sa = cat @ proj_w + b ; resid = x + sa  (bf16 out)
//   6) gemm:   out = resid @ fc_w + fc_b               (f32 out)
// ---------------------------------------------------------------------------

#define SEQ        2048
#define N_EMBD     2048
#define N_HEAD     16
#define HEAD_SIZE  128
#define NUM_TOKENS 32000

typedef __attribute__((ext_vector_type(16))) __bf16 v16bf;
typedef __attribute__((ext_vector_type(8)))  float  v8f;

union Frag16 {                 // one WMMA 16-bit A/B fragment (8 VGPRs)
    v16bf          bf;
    uint4          u[2];
    unsigned short h[16];
};

__device__ __forceinline__ unsigned short f2bf(float f) {
    unsigned int u = __float_as_uint(f);
    u += 0x7FFFu + ((u >> 16) & 1u);          // round-to-nearest-even
    return (unsigned short)(u >> 16);
}

// ---------------------------------------------------------------------------
// 1) Embedding gather + positional encoding. Writes f32 x and bf16 xb.
// ---------------------------------------------------------------------------
__global__ __launch_bounds__(256)
void k_embed(const int* __restrict__ ids, const float* __restrict__ we,
             const float* __restrict__ pe, float* __restrict__ x,
             unsigned short* __restrict__ xb) {
    const int t  = blockIdx.x;
    const int id = ids[t];
    const float4* wr = (const float4*)(we + (long)id * N_EMBD);
    const float4* pr = (const float4*)(pe + (long)t  * N_EMBD);
    float4*       xr = (float4*)(x + (long)t * N_EMBD);
    ushort4*      br = (ushort4*)(xb + (long)t * N_EMBD);
    for (int i = threadIdx.x; i < N_EMBD / 4; i += 256) {
        float4 a = wr[i], b = pr[i];
        float4 v = make_float4(a.x + b.x, a.y + b.y, a.z + b.z, a.w + b.w);
        xr[i] = v;
        br[i] = make_ushort4(f2bf(v.x), f2bf(v.y), f2bf(v.z), f2bf(v.w));
    }
}

// ---------------------------------------------------------------------------
// 2a) f32 [R][C] -> bf16 transposed [C][R], batched; 32x32 LDS tiles,
//     coalesced on both sides.
// ---------------------------------------------------------------------------
__global__ __launch_bounds__(256)
void k_cvtT_bf16(const float* __restrict__ in, unsigned short* __restrict__ out,
                 int R, int C) {
    __shared__ unsigned short t[32][33];
    const long zoff = (long)blockIdx.z * R * C;
    in  += zoff;
    out += zoff;
    const int cb = blockIdx.x * 32, rb = blockIdx.y * 32;
    const int tx = threadIdx.x, ty = threadIdx.y;     // 32 x 8
#pragma unroll
    for (int i = 0; i < 4; ++i) {
        const int r = rb + ty + i * 8;
        t[tx][ty + i * 8] = f2bf(in[(long)r * C + cb + tx]);
    }
    __syncthreads();
#pragma unroll
    for (int i = 0; i < 4; ++i) {
        const int c = cb + ty + i * 8;
        out[(long)c * R + rb + tx] = t[ty + i * 8][tx];
    }
}

// 2b) bf16 [R][C] -> bf16 transposed [C][R], batched (for V -> V^T).
__global__ __launch_bounds__(256)
void k_transT_bf16(const unsigned short* __restrict__ in,
                   unsigned short* __restrict__ out, int R, int C) {
    __shared__ unsigned short t[32][33];
    const long zoff = (long)blockIdx.z * R * C;
    in  += zoff;
    out += zoff;
    const int cb = blockIdx.x * 32, rb = blockIdx.y * 32;
    const int tx = threadIdx.x, ty = threadIdx.y;
#pragma unroll
    for (int i = 0; i < 4; ++i) {
        const int r = rb + ty + i * 8;
        t[tx][ty + i * 8] = in[(long)r * C + cb + tx];
    }
    __syncthreads();
#pragma unroll
    for (int i = 0; i < 4; ++i) {
        const int c = cb + ty + i * 8;
        out[(long)c * R + rb + tx] = t[ty + i * 8][tx];
    }
}

// ---------------------------------------------------------------------------
// 3/5/6) bf16 WMMA GEMM with pre-transposed B:
//   C[M,N] = A[M,K] * Bt[N,K]^T (+bias)(+resid)
//   BM=128 BN=128 BK=32, 256 threads = 8 waves; each wave: 64x32 = 4x2 WMMAs.
//   Double-buffered LDS staging, all-b128 copies both matrices.
//   Global loads for tile t+1 issue before the WMMAs of tile t.
// ---------------------------------------------------------------------------
template <bool HAS_BIAS, bool HAS_RESID, bool OUT_F, bool OUT_B>
__global__ __launch_bounds__(256)
void k_gemm_bf16(const unsigned short* __restrict__ A,
                 const unsigned short* __restrict__ Bt,
                 float* __restrict__ Cf, unsigned short* __restrict__ Cb,
                 const float* __restrict__ bias, const float* __restrict__ resid,
                 int M, int N, int K, long sA, long sB, long sC) {
    __shared__ __align__(16) unsigned short smA[2][128 * 40];   // [m][k] pad->40
    __shared__ __align__(16) unsigned short smB[2][128 * 40];   // [n][k] pad->40

    A  += (long)blockIdx.z * sA;
    Bt += (long)blockIdx.z * sB;
    const long cbase = (long)blockIdx.z * sC;

    const int bm = blockIdx.y * 128, bn = blockIdx.x * 128;
    const int tid  = threadIdx.x;
    const int lane = tid & 31, wid = tid >> 5;
    const int wm = wid >> 2, wn = wid & 3;        // 2 x 4 wave grid
    const int lr = lane & 15, hh = lane >> 4;

    // per-thread staging coordinates (identical pattern for A and Bt)
    const int r0 = (tid + 0)   >> 2, g0 = (tid + 0)   & 3;
    const int r1 = (tid + 256) >> 2, g1 = (tid + 256) & 3;

    const v8f vz = {0.f, 0.f, 0.f, 0.f, 0.f, 0.f, 0.f, 0.f};
    v8f acc[4][2];
#pragma unroll
    for (int mt = 0; mt < 4; ++mt)
#pragma unroll
        for (int nt = 0; nt < 2; ++nt) acc[mt][nt] = vz;

    uint4 ra0, ra1, rb0, rb1;
    auto gload = [&](int kk) {
        ra0 = *(const uint4*)(A  + (long)(bm + r0) * K + kk + g0 * 8);
        ra1 = *(const uint4*)(A  + (long)(bm + r1) * K + kk + g1 * 8);
        rb0 = *(const uint4*)(Bt + (long)(bn + r0) * K + kk + g0 * 8);
        rb1 = *(const uint4*)(Bt + (long)(bn + r1) * K + kk + g1 * 8);
    };
    auto sstore = [&](int p) {
        *(uint4*)&smA[p][r0 * 40 + g0 * 8] = ra0;
        *(uint4*)&smA[p][r1 * 40 + g1 * 8] = ra1;
        *(uint4*)&smB[p][r0 * 40 + g0 * 8] = rb0;
        *(uint4*)&smB[p][r1 * 40 + g1 * 8] = rb1;
    };

    const int nkt = K / 32;
    gload(0);
    sstore(0);
    for (int t = 0; t < nkt; ++t) {
        __syncthreads();
        const int cur  = t & 1;
        const bool more = (t + 1) < nkt;
        if (more) gload((t + 1) * 32);          // overlap HBM/L2 with WMMAs

        Frag16 af[4], bf2[2];
#pragma unroll
        for (int mt = 0; mt < 4; ++mt) {                     // A: K 0-7|16-23 per half
            const unsigned short* p = &smA[cur][(wm * 64 + mt * 16 + lr) * 40 + hh * 8];
            af[mt].u[0] = *(const uint4*)p;
            af[mt].u[1] = *(const uint4*)(p + 16);
        }
#pragma unroll
        for (int nt = 0; nt < 2; ++nt) {                     // B: K 0-7|16-23 per half
            const unsigned short* p = &smB[cur][(wn * 32 + nt * 16 + lr) * 40 + hh * 8];
            bf2[nt].u[0] = *(const uint4*)p;
            bf2[nt].u[1] = *(const uint4*)(p + 16);
        }
#pragma unroll
        for (int mt = 0; mt < 4; ++mt)
#pragma unroll
            for (int nt = 0; nt < 2; ++nt)
                acc[mt][nt] = __builtin_amdgcn_wmma_f32_16x16x32_bf16(
                    false, af[mt].bf, false, bf2[nt].bf, (short)0, acc[mt][nt],
                    false, false);

        if (more) sstore((t + 1) & 1);          // fill other buffer; no extra barrier
    }

    // epilogue: C layout M = j + 8*(lane/16), N = lane%16 — branch-free
#pragma unroll
    for (int mt = 0; mt < 4; ++mt)
#pragma unroll
        for (int nt = 0; nt < 2; ++nt) {
            const int ng = bn + wn * 32 + nt * 16 + lr;
            const int mg0 = bm + wm * 64 + mt * 16 + hh * 8;
            float bv = 0.f;
            if (HAS_BIAS) bv = bias[ng];
            float r[8];
            if (HAS_RESID) {
#pragma unroll
                for (int j = 0; j < 8; ++j)
                    r[j] = resid[(long)(mg0 + j) * N + ng];   // loads batched
            }
#pragma unroll
            for (int j = 0; j < 8; ++j) {
                float vv = acc[mt][nt][j] + bv;
                if (HAS_RESID) vv += r[j];
                const long off = cbase + (long)(mg0 + j) * N + ng;
                if (OUT_F) Cf[off] = vv;
                if (OUT_B) Cb[off] = f2bf(vv);
            }
        }
}

// ---------------------------------------------------------------------------
// 4) Flash attention, one head x one 128-row query block per workgroup.
//    8 waves, 16 query rows each. Key blocks of 64. All-WMMA matmuls.
//    V supplied pre-transposed [h][hs][t] so all staging is b128 copies.
// ---------------------------------------------------------------------------
__global__ __launch_bounds__(256)
void k_attention(const unsigned short* __restrict__ q,
                 const unsigned short* __restrict__ k,
                 const unsigned short* __restrict__ vt,
                 unsigned short* __restrict__ cat) {
    __shared__ __align__(16) unsigned short sK[64 * 136];     // [key][d]
    __shared__ __align__(16) unsigned short sV[128 * 72];     // [d][key] (V^T)
    __shared__ __align__(16) unsigned short sP[8 * 16 * 72];  // per-wave P stage

    const int blk = blockIdx.x, h = blockIdx.y;
    const int tid = threadIdx.x, lane = tid & 31, wid = tid >> 5;
    const int lr = lane & 15, hh = lane >> 4;
    const float scale = 0.08838834764831845f;                 // 1/sqrt(128)

    const long hb = (long)h * SEQ * HEAD_SIZE;
    const int  qrow = blk * 128 + wid * 16 + lr;              // this lane's Q row

    // Q fragments pinned in registers: A layout, K dim = 128 -> 4 frags
    Frag16 qf[4];
#pragma unroll
    for (int kt = 0; kt < 4; ++kt) {
        const unsigned short* p = q + hb + (long)qrow * HEAD_SIZE + kt * 32 + hh * 8;
        qf[kt].u[0] = *(const uint4*)p;
        qf[kt].u[1] = *(const uint4*)(p + 16);
    }

    const v8f vz = {0.f, 0.f, 0.f, 0.f, 0.f, 0.f, 0.f, 0.f};
    v8f o[8];
#pragma unroll
    for (int ot = 0; ot < 8; ++ot) o[ot] = vz;
    float rmax[8], rsum[8];
#pragma unroll
    for (int j = 0; j < 8; ++j) { rmax[j] = -3.0e38f; rsum[j] = 0.f; }

    const int nkb = 2 * blk + 2;                              // causal bound
    for (int kb = 0; kb < nkb; ++kb) {
        __syncthreads();                                      // tiles free to reuse
        // load K tile [64][128] row-major (b128)
#pragma unroll
        for (int i = 0; i < 4; ++i) {
            int idx = tid + i * 256;
            int key = idx >> 4, g = idx & 15;
            *(uint4*)&sK[key * 136 + g * 8] =
                *(const uint4*)(k + hb + (long)(kb * 64 + key) * HEAD_SIZE + g * 8);
        }
        // load V^T tile [128 d][64 key] row-major (b128, pre-transposed source)
#pragma unroll
        for (int i = 0; i < 4; ++i) {
            int idx = tid + i * 256;
            int d = idx >> 3, g = idx & 7;
            *(uint4*)&sV[d * 72 + g * 8] =
                *(const uint4*)(vt + hb + (long)d * SEQ + kb * 64 + g * 8);
        }
        __syncthreads();

        // S = Q * K^T  (4 key subtiles x 4 k-steps)
        v8f s[4];
#pragma unroll
        for (int nt = 0; nt < 4; ++nt) {
            s[nt] = vz;
#pragma unroll
            for (int kt = 0; kt < 4; ++kt) {
                Frag16 bf2;
                const unsigned short* p =
                    &sK[(nt * 16 + lr) * 136 + kt * 32 + hh * 8];
                bf2.u[0] = *(const uint4*)p;
                bf2.u[1] = *(const uint4*)(p + 16);
                s[nt] = __builtin_amdgcn_wmma_f32_16x16x32_bf16(
                    false, qf[kt].bf, false, bf2.bf, (short)0, s[nt], false, false);
            }
        }

        // scale + causal mask (diagonal region only)
        const bool diag = (kb >= 2 * blk);
#pragma unroll
        for (int nt = 0; nt < 4; ++nt) {
            const int c = kb * 64 + nt * 16 + lr;
#pragma unroll
            for (int j = 0; j < 8; ++j) {
                const int r = blk * 128 + wid * 16 + j + hh * 8;
                float sv = s[nt][j] * scale;
                if (diag && c > r) sv = -1.0e30f;
                s[nt][j] = sv;
            }
        }

        // online softmax: row-max / row-sum reduced across 16-lane N groups
        float ml[8], ls[8], corr[8];
#pragma unroll
        for (int j = 0; j < 8; ++j)
            ml[j] = fmaxf(fmaxf(s[0][j], s[1][j]), fmaxf(s[2][j], s[3][j]));
#pragma unroll
        for (int j = 0; j < 8; ++j) {
#pragma unroll
            for (int m = 1; m <= 8; m <<= 1)
                ml[j] = fmaxf(ml[j], __shfl_xor(ml[j], m, 32));
            const float nm = fmaxf(rmax[j], ml[j]);
            corr[j] = __expf(rmax[j] - nm);
            rmax[j] = nm;
            ls[j] = 0.f;
        }
#pragma unroll
        for (int nt = 0; nt < 4; ++nt)
#pragma unroll
            for (int j = 0; j < 8; ++j) {
                const float p = __expf(s[nt][j] - rmax[j]);
                s[nt][j] = p;
                ls[j] += p;
            }
#pragma unroll
        for (int j = 0; j < 8; ++j) {
#pragma unroll
            for (int m = 1; m <= 8; m <<= 1) ls[j] += __shfl_xor(ls[j], m, 32);
            rsum[j] = rsum[j] * corr[j] + ls[j];
        }
#pragma unroll
        for (int ot = 0; ot < 8; ++ot)
#pragma unroll
            for (int j = 0; j < 8; ++j) o[ot][j] *= corr[j];

        // restage P (C layout) -> A layout via per-wave LDS region
        unsigned short* pw = &sP[wid * 16 * 72];
#pragma unroll
        for (int nt = 0; nt < 4; ++nt)
#pragma unroll
            for (int j = 0; j < 8; ++j)
                pw[(j + hh * 8) * 72 + nt * 16 + lr] = f2bf(s[nt][j]);
        __syncthreads();

        Frag16 pf[2];
#pragma unroll
        for (int kt = 0; kt < 2; ++kt) {
            const unsigned short* p = pw + lr * 72 + kt * 32 + hh * 8;
            pf[kt].u[0] = *(const uint4*)p;
            pf[kt].u[1] = *(const uint4*)(p + 16);
        }
        // O += P * V   (8 output subtiles over hs=128, 2 k-steps over keys)
#pragma unroll
        for (int ot = 0; ot < 8; ++ot)
#pragma unroll
            for (int kt = 0; kt < 2; ++kt) {
                Frag16 vf;
                const unsigned short* p =
                    &sV[(ot * 16 + lr) * 72 + kt * 32 + hh * 8];
                vf.u[0] = *(const uint4*)p;
                vf.u[1] = *(const uint4*)(p + 16);
                o[ot] = __builtin_amdgcn_wmma_f32_16x16x32_bf16(
                    false, pf[kt].bf, false, vf.bf, (short)0, o[ot], false, false);
            }
    }

    // normalize + write concatenated heads layout: cat[t][h*128 + d]
    float inv[8];
#pragma unroll
    for (int j = 0; j < 8; ++j) inv[j] = 1.0f / rsum[j];
#pragma unroll
    for (int ot = 0; ot < 8; ++ot) {
        const int d = ot * 16 + lr;
#pragma unroll
        for (int j = 0; j < 8; ++j) {
            const int r = blk * 128 + wid * 16 + j + hh * 8;
            cat[(long)r * N_EMBD + h * HEAD_SIZE + d] = f2bf(o[ot][j] * inv[j]);
        }
    }
}

// ---------------------------------------------------------------------------
// Host launcher
// ---------------------------------------------------------------------------
extern "C" void kernel_launch(void* const* d_in, const int* in_sizes, int n_in,
                              void* d_out, int out_size, void* d_ws, size_t ws_size,
                              hipStream_t stream) {
    const int*   ids    = (const int*)  d_in[0];
    const float* we     = (const float*)d_in[1];
    const float* pe     = (const float*)d_in[2];
    const float* Wq     = (const float*)d_in[3];
    const float* Wk     = (const float*)d_in[4];
    const float* Wv     = (const float*)d_in[5];
    const float* proj_w = (const float*)d_in[6];
    const float* proj_b = (const float*)d_in[7];
    const float* fc_w   = (const float*)d_in[8];
    const float* fc_b   = (const float*)d_in[9];
    float* out = (float*)d_out;

    // workspace carve (256B aligned)
    unsigned char* p = (unsigned char*)d_ws;
    auto carve = [&](size_t bytes) {
        void* r = (void*)p;
        p += (bytes + 255) & ~(size_t)255;
        return r;
    };
    const long Ewq = (long)N_HEAD * N_EMBD * HEAD_SIZE;       // 4.19M
    const long Exx = (long)SEQ * N_EMBD;                      // 4.19M
    const long Efw = (long)N_EMBD * NUM_TOKENS;               // 65.5M

    float*          x      = (float*)         carve(Exx * 4);
    unsigned short* xb     = (unsigned short*) carve(Exx * 2);
    unsigned short* wqbT   = (unsigned short*) carve(Ewq * 2);  // [h][hs][D]
    unsigned short* wkbT   = (unsigned short*) carve(Ewq * 2);
    unsigned short* wvbT   = (unsigned short*) carve(Ewq * 2);
    unsigned short* pwbT   = (unsigned short*) carve(Exx * 2);  // [N][K]
    unsigned short* fwbT   = (unsigned short*) carve(Efw * 2);  // [32000][2048]
    unsigned short* qb     = (unsigned short*) carve(Ewq * 2);  // [h][t][hs]
    unsigned short* kb     = (unsigned short*) carve(Ewq * 2);
    unsigned short* vb     = (unsigned short*) carve(Ewq * 2);
    unsigned short* vtb    = (unsigned short*) carve(Ewq * 2);  // [h][hs][t]
    unsigned short* catb   = (unsigned short*) carve(Exx * 2);
    unsigned short* residb = (unsigned short*) carve(Exx * 2);

    // 1) embed
    k_embed<<<SEQ, 256, 0, stream>>>(ids, we, pe, x, xb);

    // 2) transpose-convert all weights to bf16 [N][K]
    dim3 tb(32, 8);
    k_cvtT_bf16<<<dim3(HEAD_SIZE / 32, N_EMBD / 32, N_HEAD), tb, 0, stream>>>(
        Wq, wqbT, N_EMBD, HEAD_SIZE);
    k_cvtT_bf16<<<dim3(HEAD_SIZE / 32, N_EMBD / 32, N_HEAD), tb, 0, stream>>>(
        Wk, wkbT, N_EMBD, HEAD_SIZE);
    k_cvtT_bf16<<<dim3(HEAD_SIZE / 32, N_EMBD / 32, N_HEAD), tb, 0, stream>>>(
        Wv, wvbT, N_EMBD, HEAD_SIZE);
    k_cvtT_bf16<<<dim3(N_EMBD / 32, N_EMBD / 32, 1), tb, 0, stream>>>(
        proj_w, pwbT, N_EMBD, N_EMBD);
    k_cvtT_bf16<<<dim3(NUM_TOKENS / 32, N_EMBD / 32, 1), tb, 0, stream>>>(
        fc_w, fwbT, N_EMBD, NUM_TOKENS);

    // 3) QKV: batched per head, A = xb shared, Bt = W[h]^T [128 x 2048]
    {
        dim3 g(HEAD_SIZE / 128, SEQ / 128, N_HEAD);
        long sB = (long)N_EMBD * HEAD_SIZE, sC = (long)SEQ * HEAD_SIZE;
        k_gemm_bf16<false, false, false, true><<<g, 256, 0, stream>>>(
            xb, wqbT, nullptr, qb, nullptr, nullptr, SEQ, HEAD_SIZE, N_EMBD, 0, sB, sC);
        k_gemm_bf16<false, false, false, true><<<g, 256, 0, stream>>>(
            xb, wkbT, nullptr, kb, nullptr, nullptr, SEQ, HEAD_SIZE, N_EMBD, 0, sB, sC);
        k_gemm_bf16<false, false, false, true><<<g, 256, 0, stream>>>(
            xb, wvbT, nullptr, vb, nullptr, nullptr, SEQ, HEAD_SIZE, N_EMBD, 0, sB, sC);
    }
    // v -> v^T per head for b128 V staging in attention
    k_transT_bf16<<<dim3(HEAD_SIZE / 32, SEQ / 32, N_HEAD), tb, 0, stream>>>(
        vb, vtb, SEQ, HEAD_SIZE);

    // 4) flash attention -> catb
    k_attention<<<dim3(SEQ / 128, N_HEAD), 256, 0, stream>>>(qb, kb, vtb, catb);

    // 5) proj + bias + residual -> residb (bf16)
    k_gemm_bf16<true, true, false, true><<<dim3(N_EMBD / 128, SEQ / 128, 1), 256, 0, stream>>>(
        catb, pwbT, nullptr, residb, proj_b, x, SEQ, N_EMBD, N_EMBD, 0, 0, 0);

    // 6) logits = resid @ fc_w + fc_b -> d_out (f32)
    k_gemm_bf16<true, false, true, false><<<dim3(NUM_TOKENS / 128, SEQ / 128, 1), 256, 0, stream>>>(
        residb, fwbT, out, nullptr, fc_b, nullptr, SEQ, NUM_TOKENS, N_EMBD, 0, 0, 0);
}
